// dotProductAttention_6691559047624
// MI455X (gfx1250) — matrix operand order repaired and checked by reading, not compiled, not run on hardware
//
#include <hip/hip_runtime.h>

typedef __bf16 bf16_t;
typedef __bf16 v16bf __attribute__((ext_vector_type(16)));
typedef __bf16 v8bf  __attribute__((ext_vector_type(8)));
typedef __bf16 v4bf  __attribute__((ext_vector_type(4)));
typedef __bf16 v2bf  __attribute__((ext_vector_type(2)));
typedef float  v2f   __attribute__((ext_vector_type(2)));
typedef float  v8f   __attribute__((ext_vector_type(8)));

#define SEQ    2048
#define DHEAD  64
#define NHEADS 64      // B*H = 4*16
#define KSTR   72      // padded LDS row stride (elements) for K tile   (144B, 16B-aligned)
#define VSTR   40      // padded LDS row stride for V^T tile            (80B, 16B-aligned)

// ---- f32 -> bf16: vector fptrunc; backend lowers to packed v_cvt when available ----
static __device__ __forceinline__ v2bf pk2(float a, float b) {
  v2f t; t[0] = a; t[1] = b;
  return __builtin_convertvector(t, v2bf);
}

// ---- lane^16 exchange: VALU permlane, else shuffle ----
#if __has_builtin(__builtin_amdgcn_permlanex16)
static __device__ __forceinline__ float xhalf(float x) {
  int i = __builtin_bit_cast(int, x);
  int r = __builtin_amdgcn_permlanex16(i, i, 0x76543210, 0xFEDCBA98, false, false);
  return __builtin_bit_cast(float, r);
}
#else
static __device__ __forceinline__ float xhalf(float x) { return __shfl_xor(x, 16, 32); }
#endif

static __device__ __forceinline__ v16bf cat16(v8bf lo, v8bf hi) {
  return __builtin_shufflevector(lo, hi, 0,1,2,3,4,5,6,7,8,9,10,11,12,13,14,15);
}

// A-fragment slice: two 8-element contiguous chunks from LDS (two ds_load_b128)
static __device__ __forceinline__ v16bf ldsA16(const bf16_t* p0, const bf16_t* p1) {
  v8bf lo = *(const v8bf*)p0;
  v8bf hi = *(const v8bf*)p1;
  return cat16(lo, hi);
}

// 16 scaled f32 (two runs of 8) -> bf16 B-fragment slice (packed converts)
static __device__ __forceinline__ v16bf cvt16(float4 a, float4 b, float4 c, float4 d, float s) {
  v16bf r;
  v2bf p;
  p = pk2(a.x*s, a.y*s); r[0]  = p[0]; r[1]  = p[1];
  p = pk2(a.z*s, a.w*s); r[2]  = p[0]; r[3]  = p[1];
  p = pk2(b.x*s, b.y*s); r[4]  = p[0]; r[5]  = p[1];
  p = pk2(b.z*s, b.w*s); r[6]  = p[0]; r[7]  = p[1];
  p = pk2(c.x*s, c.y*s); r[8]  = p[0]; r[9]  = p[1];
  p = pk2(c.z*s, c.w*s); r[10] = p[0]; r[11] = p[1];
  p = pk2(d.x*s, d.y*s); r[12] = p[0]; r[13] = p[1];
  p = pk2(d.z*s, d.w*s); r[14] = p[0]; r[15] = p[1];
  return r;
}

static __device__ __forceinline__ v8f wmma_bf16(v16bf a, v16bf b, v8f c) {
  return __builtin_amdgcn_wmma_f32_16x16x32_bf16(false, a, false, b, (short)0, c, false, false);
}

__global__ __launch_bounds__(128)
void fa_fwd_kernel(const float* __restrict__ Q, const float* __restrict__ K,
                   const float* __restrict__ V, float* __restrict__ O) {
  __shared__ alignas(16) bf16_t KS[32 * KSTR];   // K tile row-major (kv x 64), bf16
  __shared__ alignas(16) bf16_t VT[64 * VSTR];   // V tile transposed (64 x kv), bf16

  const int tid  = threadIdx.x;
  const int lane = tid & 31;
  const int ln   = lane & 15;   // q-column owned by this lane (N index)
  const int hl   = lane >> 4;

  const int qblocks = SEQ / 64;
  const int bh   = blockIdx.x / qblocks;
  const int qblk = blockIdx.x % qblocks;
  const long base = (long)bh * SEQ * DHEAD;
  const int q0 = qblk * 64 + (tid >> 5) * 16;

  // ---- Q as B-fragments of S^T = K Q^T (d is the WMMA K-dim; contiguous in memory) ----
  const float sc = 0.125f;  // 1/sqrt(64) folded into Q
  const float4* q4 = (const float4*)(Q + base + (long)(q0 + ln) * DHEAD);
  v16bf qb0 = cvt16(q4[2*hl + 0], q4[2*hl + 1], q4[4 + 2*hl + 0], q4[4 + 2*hl + 1], sc);                 // d = 8*hl..+7, 16+8*hl..+7
  v16bf qb1 = cvt16(q4[8 + 2*hl + 0], q4[8 + 2*hl + 1], q4[12 + 2*hl + 0], q4[12 + 2*hl + 1], sc); // d = 32+8*hl..+7, 48+8*hl..+7

  float m = -__builtin_inff();
  float l = 0.0f;
  v8f acc[4] = {v8f{}, v8f{}, v8f{}, v8f{}};

  const float* Kbh = K + base;
  const float* Vbh = V + base;

  // software-pipelined staging registers (each thread: 4 float4 of K, 4 of V)
  float4 pkr[4], pvr[4];
  {
    const float4* ks = (const float4*)Kbh;
    const float4* vs = (const float4*)Vbh;
#pragma unroll
    for (int i = 0; i < 4; ++i) { pkr[i] = ks[tid + i*128]; pvr[i] = vs[tid + i*128]; }
  }

  for (int kv0 = 0; kv0 < SEQ; kv0 += 32) {
    __syncthreads();  // previous tile fully consumed
    // ---- convert + store staged tile: K row-major, V transposed ----
#pragma unroll
    for (int i = 0; i < 4; ++i) {
      int fi  = tid + i * 128;
      int row = fi >> 4;               // kv-local 0..31
      int c4  = (fi & 15) << 2;        // d 0..60
      v2bf ka = pk2(pkr[i].x, pkr[i].y);
      v2bf kb = pk2(pkr[i].z, pkr[i].w);
      v4bf kq; kq[0] = ka[0]; kq[1] = ka[1]; kq[2] = kb[0]; kq[3] = kb[1];
      *(v4bf*)&KS[row * KSTR + c4] = kq;
      v2bf va = pk2(pvr[i].x, pvr[i].y);
      v2bf vb = pk2(pvr[i].z, pvr[i].w);
      VT[(c4 + 0) * VSTR + row] = va[0];
      VT[(c4 + 1) * VSTR + row] = va[1];
      VT[(c4 + 2) * VSTR + row] = vb[0];
      VT[(c4 + 3) * VSTR + row] = vb[1];
    }
    // ---- prefetch next tile into registers (overlaps with compute below) ----
    {
      int nxt = (kv0 + 32 < SEQ) ? (kv0 + 32) : kv0;  // branchless; last iter reloads (unused)
      const float4* ks = (const float4*)(Kbh + (long)nxt * DHEAD);
      const float4* vs = (const float4*)(Vbh + (long)nxt * DHEAD);
#pragma unroll
      for (int i = 0; i < 4; ++i) { pkr[i] = ks[tid + i*128]; pvr[i] = vs[tid + i*128]; }
    }
    __syncthreads();  // staged tile visible

    // ---- S^T = K Q^T : two 16(kv) x 16(q) tiles, 4 WMMAs ----
    const bf16_t* kr0 = &KS[ln * KSTR];          // A rows kv = 0..15
    const bf16_t* kr1 = &KS[(16 + ln) * KSTR];   // A rows kv = 16..31
    v8f t0 = {}, t1 = {};
    t0 = wmma_bf16(ldsA16(kr0 + 8*hl,      kr0 + 16 + 8*hl), qb0, t0);
    t0 = wmma_bf16(ldsA16(kr0 + 32 + 8*hl, kr0 + 48 + 8*hl), qb1, t0);
    t1 = wmma_bf16(ldsA16(kr1 + 8*hl,      kr1 + 16 + 8*hl), qb0, t1);
    t1 = wmma_bf16(ldsA16(kr1 + 32 + 8*hl, kr1 + 48 + 8*hl), qb1, t1);

    // ---- online softmax: each lane owns one q row (N = ln); kv split across lane halves ----
    float mx = -__builtin_inff();
#pragma unroll
    for (int v = 0; v < 8; ++v) mx = fmaxf(mx, fmaxf(t0[v], t1[v]));
    mx = fmaxf(mx, xhalf(mx));                   // single cross-half exchange (v_permlanex16)
    float mn = fmaxf(m, mx);
    float p0[8], p1[8];
    float rs = 0.0f;
#pragma unroll
    for (int v = 0; v < 8; ++v) {
      p0[v] = __expf(t0[v] - mn);
      p1[v] = __expf(t1[v] - mn);
      rs += p0[v] + p1[v];
    }
    rs += xhalf(rs);
    float corr = __expf(m - mn);                 // exp(-inf)=0 covers the first tile
    m = mn;
    l = l * corr + rs;
#pragma unroll
    for (int j = 0; j < 4; ++j) acc[j] *= corr;

    // ---- build P^T as B-fragment in registers (8 cross-half exchanges + packed cvt) ----
    // half0 lane holds kv {0..7}u{16..23}; half1 holds {8..15}u{24..31}.
    // B-frag wants: half0 -> kv 0..15, half1 -> kv 16..31.
    float lo[8], hi[8];
#pragma unroll
    for (int v = 0; v < 8; ++v) {
      float send = hl ? p0[v] : p1[v];
      float recv = xhalf(send);
      lo[v] = hl ? recv  : p0[v];   // element v     : kv = 16*hl + v
      hi[v] = hl ? p1[v] : recv;    // element 8 + v : kv = 16*hl + 8 + v
    }
    v16bf pb;
#pragma unroll
    for (int v = 0; v < 8; v += 2) {
      v2bf a = pk2(lo[v], lo[v+1]); pb[v]     = a[0]; pb[v+1]     = a[1];
      v2bf b = pk2(hi[v], hi[v+1]); pb[8 + v] = b[0]; pb[8 + v+1] = b[1];
    }

    // ---- O^T += V^T P^T : 4 d-subtiles, 4 WMMAs ----
#pragma unroll
    for (int j = 0; j < 4; ++j) {
      const bf16_t* vr = &VT[(16*j + ln) * VSTR];
      acc[j] = wmma_bf16(ldsA16(vr + 16*hl, vr + 16*hl + 8), pb, acc[j]);
    }
  }

  // ---- epilogue: normalize, store fp32 as float4 (d contiguous in acc slots) ----
  float inv = 1.0f / l;
  float* orow = O + base + (long)(q0 + ln) * DHEAD;
#pragma unroll
  for (int j = 0; j < 4; ++j) {
    float4 o0, o1;
    o0.x = acc[j][0]*inv; o0.y = acc[j][1]*inv; o0.z = acc[j][2]*inv; o0.w = acc[j][3]*inv;
    o1.x = acc[j][4]*inv; o1.y = acc[j][5]*inv; o1.z = acc[j][6]*inv; o1.w = acc[j][7]*inv;
    float* p = orow + 16*j + 8*hl;   // d = 16j + 8*hl + (0..7)
    *(float4*)(p)     = o0;
    *(float4*)(p + 4) = o1;
  }
}

extern "C" void kernel_launch(void* const* d_in, const int* in_sizes, int n_in,
                              void* d_out, int out_size, void* d_ws, size_t ws_size,
                              hipStream_t stream) {
  const float* Q = (const float*)d_in[0];
  const float* K = (const float*)d_in[1];
  const float* V = (const float*)d_in[2];
  float* O = (float*)d_out;
  (void)in_sizes; (void)n_in; (void)out_size; (void)d_ws; (void)ws_size;

  dim3 grid(NHEADS * (SEQ / 64));   // 64 heads * 32 q-blocks = 2048 blocks
  dim3 block(128);                  // 4 waves; each wave owns a 16-row Q tile
  fa_fwd_kernel<<<grid, block, 0, stream>>>(Q, K, V, O);
}
